// Net_17025250361809
// MI455X (gfx1250) — matrix operand order repaired and checked by reading, not compiled, not run on hardware
//
#include <hip/hip_runtime.h>
#include <hip/hip_bf16.h>

typedef _Float16 v16h __attribute__((ext_vector_type(16)));
typedef float    v8f  __attribute__((ext_vector_type(8)));

#define NB      8192
#define NODES   54
#define EPER    144
#define ETOT    (NB * EPER)

union HU { uint4 u4[2]; v16h h; };

__device__ __forceinline__ int a_kmap(int i, int hf) {
    // 16-bit A-matrix 16x32 per-lane element -> K index (contiguous: K = hf*16 + i, matches B packing)
    return hf * 16 + i;
}

__device__ __forceinline__ unsigned lds_off(const void* p) {
    // generic LDS pointer: low 32 bits are the wave-relative LDS byte offset
    return (unsigned)(unsigned long long)p;
}

// ---------------------------------------------------------------------------
// Kernel 0: repack weights to f16 in WMMA B-operand lane order.
// B 32x16 tile: lane L holds col N=L%16; element i -> K = (L/16)*16 + i.
// ---------------------------------------------------------------------------
__global__ void prep_kernel(const float* __restrict__ W1_rel,
                            const float* __restrict__ W1_root,
                            const float* __restrict__ W2_rel,
                            const float* __restrict__ W2_root,
                            const float* __restrict__ O1W,
                            _Float16* __restrict__ b1p,   // [4 ct][32 lane][16]
                            _Float16* __restrict__ b2p,   // [4 ks][32 lane][16]
                            _Float16* __restrict__ O1p)   // [8 ks][8 ct][32][16]
{
    int t = blockIdx.x * 256 + threadIdx.x;
    if (t < 2048) {
        int ct = t >> 9, lane = (t >> 4) & 31, i = t & 15;
        int hf = lane >> 4, n = ct * 16 + (lane & 15);
        int k = hf * 16 + i;                    // K = 32 = [W1_rel(16) ; W1_root(16)]
        float v = (k < 16) ? W1_rel[k * 64 + n] : W1_root[(k - 16) * 64 + n];
        b1p[t] = (_Float16)v;
    } else if (t < 4096) {
        int u = t - 2048;
        int ks = u >> 9, lane = (u >> 4) & 31, i = u & 15;
        int hf = lane >> 4, n = lane & 15;
        int k = ks * 32 + hf * 16 + i;          // K = 128 = [W2_rel(64) ; W2_root(64)]
        float v = 0.f;
        if (n < 4) v = (k < 64) ? W2_rel[k * 4 + n] : W2_root[(k - 64) * 4 + n];
        b2p[u] = (_Float16)v;
    } else if (t < 4096 + 32768) {
        int u = t - 4096;
        int ks = u >> 12, ct = (u >> 9) & 7, lane = (u >> 4) & 31, i = u & 15;
        int hf = lane >> 4, n = ct * 16 + (lane & 15);
        int k = ks * 32 + hf * 16 + i;          // K padded 232 -> 256
        O1p[u] = (_Float16)((k < 232) ? O1W[k * 128 + n] : 0.f);
    }
}

// ---------------------------------------------------------------------------
// Kernel 1: fused 2-layer GraphConv + global MLP, one block per graph.
// Edge aggregation is done as dense Adj-GEMM on the matrix unit:
//   Adj[dst][src] += w  (144 LDS atomics), agg = Adj @ feat  (WMMA)
// Writes per-graph row of 256 f16: [embeds 216 | g 16 | zeros 24]
// ---------------------------------------------------------------------------
__global__ void __launch_bounds__(256)
gconv_kernel(const float* __restrict__ x, const int* __restrict__ eidx,
             const float* __restrict__ eattr, const float* __restrict__ gf,
             const float* __restrict__ b1, const float* __restrict__ b2,
             const float* __restrict__ G1W, const float* __restrict__ G1b,
             const float* __restrict__ G2W, const float* __restrict__ G2b,
             const float* __restrict__ G3W, const float* __restrict__ G3b,
             const _Float16* __restrict__ b1p, const _Float16* __restrict__ b2p,
             _Float16* __restrict__ scratchA)
{
    __shared__ float    sxf[NODES * 16];     // x tile f32 (async-filled)
    __shared__ float    adjF[64 * 64];       // adjacency accum f32
    __shared__ _Float16 adjH[64 * 64];       // adjacency f16 (WMMA A)
    __shared__ _Float16 xT[16 * 64];         // x^T f16 (WMMA B: [dim][node])
    __shared__ _Float16 sA1[64 * 32];        // A1 = [agg1 | x] f16
    __shared__ _Float16 h1T[64 * 64];        // h1^T f16 (WMMA B for agg2)
    __shared__ _Float16 sA2[64 * 128];       // A2 = [agg2 | h1] f16
    __shared__ float    sg1[8], sg2[8];

    const int g = blockIdx.x;
    const int tid = threadIdx.x;
    const int lane = tid & 31, w = tid >> 5;
    const int hf = lane >> 4, ln = lane & 15;

    // async copy x tile (54*16 f32 = 216 x b128) global -> LDS
    if (tid < 216) {
        unsigned ldsa = lds_off(sxf) + (unsigned)tid * 16u;
        unsigned voff = (unsigned)tid * 16u;
        unsigned long long gbase = (unsigned long long)(x + (size_t)g * NODES * 16);
        asm volatile("global_load_async_to_lds_b128 %0, %1, %2"
                     :: "v"(ldsa), "v"(voff), "s"(gbase) : "memory");
    }
    for (int idx = tid; idx < 64 * 64; idx += 256) adjF[idx] = 0.f;
    __syncthreads();

    // build weighted adjacency: one atomic per edge (handles duplicate edges)
    if (tid < EPER) {
        int s = eidx[g * EPER + tid] - g * NODES;
        int d = eidx[ETOT + g * EPER + tid] - g * NODES;
        atomicAdd(&adjF[d * 64 + s], eattr[g * EPER + tid]);
    }
    asm volatile("s_wait_asynccnt 0x0" ::: "memory");
    __syncthreads();

    for (int idx = tid; idx < 64 * 64; idx += 256) adjH[idx] = (_Float16)adjF[idx];
    for (int idx = tid; idx < 16 * 64; idx += 256) {
        int node = idx & 63, d = idx >> 6;
        _Float16 hv = (node < NODES) ? (_Float16)sxf[node * 16 + d] : (_Float16)0.f;
        xT[d * 64 + node] = hv;          // B-operand source for agg1
        sA1[node * 32 + 16 + d] = hv;    // right half of A1
    }
    __syncthreads();

    // agg1 = Adj[64x64] @ x[64x16]  (waves 0..3, one 16x16 tile each, K=64)
    if (w < 4) {
        v8f c = {};
#pragma unroll
        for (int ks = 0; ks < 2; ks++) {
            v16h a, b;
#pragma unroll
            for (int i = 0; i < 16; i++) {
                a[i] = adjH[(w * 16 + ln) * 64 + 32 * ks + a_kmap(i, hf)];
                b[i] = xT[ln * 64 + 32 * ks + hf * 16 + i];
            }
            c = __builtin_amdgcn_wmma_f32_16x16x32_f16(false, a, false, b, (short)0, c, false, false);
        }
#pragma unroll
        for (int r = 0; r < 8; r++)
            sA1[(w * 16 + r + hf * 8) * 32 + ln] = (_Float16)c[r];
    }
    __syncthreads();

    // layer-1: H1[64][64] = A1[64][32] @ B1[32][64] + b1, ReLU; 16 tiles, 2/wave
    for (int t = w; t < 16; t += 8) {
        int rt = t >> 2, ct = t & 3;
        v16h a, b; v8f c = {};
#pragma unroll
        for (int i = 0; i < 16; i++) a[i] = sA1[(rt * 16 + ln) * 32 + a_kmap(i, hf)];
        HU ub; const uint4* bp = (const uint4*)b1p + (ct * 32 + lane) * 2;
        ub.u4[0] = bp[0]; ub.u4[1] = bp[1]; b = ub.h;
        c = __builtin_amdgcn_wmma_f32_16x16x32_f16(false, a, false, b, (short)0, c, false, false);
#pragma unroll
        for (int r = 0; r < 8; r++) {
            int row = rt * 16 + r + hf * 8, col = ct * 16 + ln;
            float v = c[r] + b1[col]; v = v > 0.f ? v : 0.f;
            h1T[col * 64 + row] = (_Float16)v;       // B-operand source for agg2
            sA2[row * 128 + 64 + col] = (_Float16)v; // right half of A2
        }
    }
    __syncthreads();

    // agg2 = Adj[64x64] @ h1[64x64]  (16 tiles, 2/wave, K=64)
    for (int t = w; t < 16; t += 8) {
        int rt = t >> 2, ct = t & 3;
        v8f c = {};
#pragma unroll
        for (int ks = 0; ks < 2; ks++) {
            v16h a, b;
#pragma unroll
            for (int i = 0; i < 16; i++) {
                a[i] = adjH[(rt * 16 + ln) * 64 + 32 * ks + a_kmap(i, hf)];
                b[i] = h1T[(ct * 16 + ln) * 64 + 32 * ks + hf * 16 + i];
            }
            c = __builtin_amdgcn_wmma_f32_16x16x32_f16(false, a, false, b, (short)0, c, false, false);
        }
#pragma unroll
        for (int r = 0; r < 8; r++)
            sA2[(rt * 16 + r + hf * 8) * 128 + ct * 16 + ln] = (_Float16)c[r];
    }
    __syncthreads();

    // layer-2: H2[64][16] = A2[64][128] @ B2[128][16]; waves 0..3, rt = w
    if (w < 4) {
        v8f c = {};
        for (int ks = 0; ks < 4; ks++) {
            v16h a, b;
#pragma unroll
            for (int i = 0; i < 16; i++)
                a[i] = sA2[(w * 16 + ln) * 128 + 32 * ks + a_kmap(i, hf)];
            HU ub; const uint4* bp = (const uint4*)b2p + (ks * 32 + lane) * 2;
            ub.u4[0] = bp[0]; ub.u4[1] = bp[1]; b = ub.h;
            c = __builtin_amdgcn_wmma_f32_16x16x32_f16(false, a, false, b, (short)0, c, false, false);
        }
#pragma unroll
        for (int r = 0; r < 8; r++) {
            int row = w * 16 + r + hf * 8, col = ln;
            if (row < NODES && col < 4) {
                float v = c[r] + b2[col]; v = v > 0.f ? v : 0.f;
                scratchA[g * 256 + row * 4 + col] = (_Float16)v;  // embed idx = node*4+col
            }
        }
    }
    __syncthreads();

    // global MLP 16 -> 8 -> 8 -> 16 (tiny, scalar)
    if (tid < 8) {
        float s = G1b[tid];
        for (int i = 0; i < 16; i++) s += gf[g * 16 + i] * G1W[i * 8 + tid];
        sg1[tid] = s > 0.f ? s : 0.f;
    }
    __syncthreads();
    if (tid < 8) {
        float s = G2b[tid];
        for (int i = 0; i < 8; i++) s += sg1[i] * G2W[i * 8 + tid];
        sg2[tid] = s > 0.f ? s : 0.f;
    }
    __syncthreads();
    if (tid < 16) {
        float s = G3b[tid];
        for (int i = 0; i < 8; i++) s += sg2[i] * G3W[i * 16 + tid];
        s = s > 0.f ? s : 0.f;
        scratchA[g * 256 + 216 + tid] = (_Float16)s;
    }
    if (tid >= 232) scratchA[g * 256 + tid] = (_Float16)0.f;   // K pad 232->256
}

// ---------------------------------------------------------------------------
// Kernel 2: head GEMM [8192,256]@[256,128] + O1 bias/ReLU + O2 dot + sigmoid.
// One block = 16 graphs; wave w owns output col-tile w; 8 WMMA k-steps.
// ---------------------------------------------------------------------------
__global__ void __launch_bounds__(256)
head_kernel(const _Float16* __restrict__ scratchA, const _Float16* __restrict__ O1p,
            const float* __restrict__ O1b, const float* __restrict__ O2W,
            const float* __restrict__ O2b, float* __restrict__ out)
{
    __shared__ _Float16 sA[16 * 256];
    __shared__ float    part[16][8];
    const int tid = threadIdx.x;
    const int lane = tid & 31, w = tid >> 5;
    const int hf = lane >> 4, ln = lane & 15;
    const int g0 = blockIdx.x * 16;

    // async copy A tile (16 graphs x 256 f16 = 512 x b128) global -> LDS
    {
        unsigned long long gbase = (unsigned long long)(scratchA + (size_t)g0 * 256);
        unsigned lbase = lds_off(sA);
        for (int idx = tid; idx < 512; idx += 256) {
            unsigned ldsa = lbase + (unsigned)idx * 16u;
            unsigned voff = (unsigned)idx * 16u;
            asm volatile("global_load_async_to_lds_b128 %0, %1, %2"
                         :: "v"(ldsa), "v"(voff), "s"(gbase) : "memory");
        }
        asm volatile("s_wait_asynccnt 0x0" ::: "memory");
    }
    __syncthreads();

    v8f c = {};
    for (int ks = 0; ks < 8; ks++) {
        v16h a, b;
#pragma unroll
        for (int i = 0; i < 16; i++) a[i] = sA[ln * 256 + 32 * ks + a_kmap(i, hf)];
        HU ub; const uint4* bp = (const uint4*)O1p + ((ks * 8 + w) * 32 + lane) * 2;
        ub.u4[0] = bp[0]; ub.u4[1] = bp[1]; b = ub.h;
        c = __builtin_amdgcn_wmma_f32_16x16x32_f16(false, a, false, b, (short)0, c, false, false);
    }

    const int col = w * 16 + ln;
    const float bias = O1b[col], w2 = O2W[col];
#pragma unroll
    for (int r = 0; r < 8; r++) {
        float v = c[r] + bias; v = v > 0.f ? v : 0.f; v *= w2;
        // reduce over the 16 cols held within each 16-lane half (rows r / r+8)
        v += __shfl_xor(v, 1, 32);
        v += __shfl_xor(v, 2, 32);
        v += __shfl_xor(v, 4, 32);
        v += __shfl_xor(v, 8, 32);
        if (ln == 0) part[r + hf * 8][w] = v;
    }
    __syncthreads();
    if (tid < 16) {
        float s = O2b[0];
        for (int j = 0; j < 8; j++) s += part[tid][j];
        out[g0 + tid] = 1.f / (1.f + __expf(-s));
    }
}

// ---------------------------------------------------------------------------
extern "C" void kernel_launch(void* const* d_in, const int* in_sizes, int n_in,
                              void* d_out, int out_size, void* d_ws, size_t ws_size,
                              hipStream_t stream)
{
    const float* x      = (const float*)d_in[0];
    const int*   eidx   = (const int*)  d_in[1];
    const float* eattr  = (const float*)d_in[2];
    const float* gf     = (const float*)d_in[3];
    const float* W1_rel = (const float*)d_in[4];
    const float* b1     = (const float*)d_in[5];
    const float* W1_root= (const float*)d_in[6];
    const float* W2_rel = (const float*)d_in[7];
    const float* b2     = (const float*)d_in[8];
    const float* W2_root= (const float*)d_in[9];
    const float* G1W = (const float*)d_in[10]; const float* G1b = (const float*)d_in[11];
    const float* G2W = (const float*)d_in[12]; const float* G2b = (const float*)d_in[13];
    const float* G3W = (const float*)d_in[14]; const float* G3b = (const float*)d_in[15];
    const float* O1W = (const float*)d_in[16]; const float* O1b = (const float*)d_in[17];
    const float* O2W = (const float*)d_in[18]; const float* O2b = (const float*)d_in[19];

    char* ws = (char*)d_ws;
    _Float16* scratchA = (_Float16*)ws;                       // 8192*256 f16 = 4 MB
    _Float16* b1p = (_Float16*)(ws + (size_t)NB * 256 * 2);   // 2048 f16
    _Float16* b2p = b1p + 2048;                               // 2048 f16
    _Float16* O1p = b2p + 2048;                               // 32768 f16

    prep_kernel<<<144, 256, 0, stream>>>(W1_rel, W1_root, W2_rel, W2_root, O1W,
                                         b1p, b2p, O1p);
    gconv_kernel<<<NB, 256, 0, stream>>>(x, eidx, eattr, gf, b1, b2,
                                         G1W, G1b, G2W, G2b, G3W, G3b,
                                         b1p, b2p, scratchA);
    head_kernel<<<NB / 16, 256, 0, stream>>>(scratchA, O1p, O1b, O2W, O2b,
                                             (float*)d_out);
}
